// TimeseriesGenerator_18330920419813
// MI455X (gfx1250) — matrix operand order repaired and checked by reading, not compiled, not run on hardware
//
#include <hip/hip_runtime.h>
#include <hip/hip_bf16.h>

typedef __bf16 bf16;
typedef __attribute__((ext_vector_type(16))) __bf16 v16bf;
typedef __attribute__((ext_vector_type(8)))  __bf16 v8bf;
typedef __attribute__((ext_vector_type(8)))  float  v8f;

#define B_   256
#define IN_  128
#define H_   256
#define G4_  1024   // 4*H
#define T_   512
#define F_   64

union BF16x16 { v16bf v; struct { v8bf lo, hi; } p; };

__device__ __forceinline__ v8f wmma_bf16(v16bf a, v16bf b, v8f c) {
  // D = A(16x32 bf16) x B(32x16 bf16) + C(16x16 f32)
  return __builtin_amdgcn_wmma_f32_16x16x32_bf16(false, a, false, b, (short)0, c,
                                                 false, false);
}

// A-fragment (16x32 bf16) from row-major LDS h[16][H_].
// Layout: lanes 0-15 -> M=lane, K in {0..15}; lanes 16-31 -> M=lane-16,
// K in {16..31}. Two 16B ds loads per lane.
__device__ __forceinline__ v16bf load_a_frag(const bf16* h, int lane, int koff) {
  int m  = lane & 15;
  int kb = (lane >> 4) << 4;                // 0 or 16
  const bf16* base = h + m * H_ + koff + kb;
  BF16x16 r;
  r.p.lo = *(const v8bf*)(base);
  r.p.hi = *(const v8bf*)(base + 8);
  return r.v;
}

// B-fragment (32x16 bf16) from pre-packed weights: Wp[k][n][lane][16] bf16,
// each lane's 16 values contiguous -> two coalesced global_load_b128 per lane.
__device__ __forceinline__ v16bf load_b_frag(const bf16* __restrict__ wp, int lane,
                                             int k, int n, int ntiles) {
  const bf16* base = wp + (size_t)((((k * ntiles + n) << 5) + lane) << 4);
  BF16x16 r;
  r.p.lo = *(const v8bf*)(base);
  r.p.hi = *(const v8bf*)(base + 8);
  return r.v;
}

__device__ __forceinline__ float sigf(float x) {
  x = fminf(fmaxf(x, -30.f), 30.f);
  return 1.f / (1.f + __expf(-x));
}
__device__ __forceinline__ float tanh_(float x) {
  x = fminf(fmaxf(x, -15.f), 15.f);
  float e = __expf(2.f * x);
  return (e - 1.f) / (e + 1.f);
}

// ---- prep: x0 = relu(inputs @ W_in + b_in), [256,256] ----
__global__ void k_x0(const float* __restrict__ in, const float* __restrict__ Win,
                     const float* __restrict__ bin, float* __restrict__ x0) {
  int idx = blockIdx.x * blockDim.x + threadIdx.x;   // 256*256
  int b = idx >> 8, hc = idx & 255;
  float acc = bin[hc];
  for (int d = 0; d < IN_; ++d) acc += in[b * IN_ + d] * Win[d * H_ + hc];
  x0[idx] = fmaxf(acc, 0.f);
}

// ---- prep: zx1 = x0 @ Wx1 + b1, [256,1024] (time-invariant: computed ONCE) ----
__global__ void k_zx1(const float* __restrict__ x0, const float* __restrict__ Wx1,
                      const float* __restrict__ b1, float* __restrict__ zx1) {
  int idx = blockIdx.x * blockDim.x + threadIdx.x;   // 256*1024
  int b = idx >> 10, c = idx & 1023;
  float acc = b1[c];
  for (int d = 0; d < H_; ++d) acc += x0[b * H_ + d] * Wx1[d * G4_ + c];
  zx1[idx] = acc;
}

// ---- prep: pack f32 weight [256][ntiles*16] into WMMA B-fragment layout (bf16) ----
// out[k][n][lane][16]; lane<16: K rows k*32..k*32+15, lane>=16: k*32+16..k*32+31; N=lane&15
__global__ void k_pack(const float* __restrict__ W, bf16* __restrict__ out, int ntiles) {
  int idx = blockIdx.x * blockDim.x + threadIdx.x;   // 8*ntiles*32
  int total = 8 * ntiles * 32;
  if (idx >= total) return;
  int lane = idx & 31;
  int tmp  = idx >> 5;
  int n = tmp % ntiles, k = tmp / ntiles;
  int Nc   = ntiles * 16;
  int col  = n * 16 + (lane & 15);
  int krow = k * 32 + ((lane >> 4) << 4);
  bf16* o = out + (size_t)idx * 16;
  for (int t = 0; t < 16; ++t) o[t] = (bf16)W[(size_t)(krow + t) * Nc + col];
}

// ---- persistent fused 2-layer LSTM + output projection ----
// wstep is 0 at runtime but opaque to the compiler: it makes the weight
// addresses loop-variant so the B-fragment loads cannot be hoisted out of the
// time loop (which previously caused catastrophic scratch spilling).
__global__ __launch_bounds__(512, 1) void lstm_fused(
    const float* __restrict__ zx1, const float* __restrict__ b2,
    const bf16* __restrict__ Wh1p0, const bf16* __restrict__ Wx2p0,
    const bf16* __restrict__ Wh2p0, const bf16* __restrict__ Woutp0,
    const float* __restrict__ bout, float* __restrict__ out, long wstep) {
  __shared__ __attribute__((aligned(16))) unsigned short h1u[16 * H_];
  __shared__ __attribute__((aligned(16))) unsigned short h2u[16 * H_];
  bf16* h1s = (bf16*)h1u;
  bf16* h2s = (bf16*)h2u;

  const int tid  = threadIdx.x;
  const int lane = tid & 31;
  const int w    = tid >> 5;             // wave 0..15 owns h-tile w
  const int b0   = blockIdx.x << 4;      // batch rows [b0, b0+16)
  const int rowb = (lane >> 4) << 3;     // WMMA C layout: M = rowb + vgpr
  const int nn   = lane & 15;            //                N = nn

  for (int i = tid; i < 16 * H_; i += 512) { h1u[i] = 0; h2u[i] = 0; }

  const int gm[4] = { w, 16 + w, 32 + w, 48 + w };   // i, f, g, o gate tiles

  // persistent per-wave registers (C operands for the k=0 WMMA of each GEMM)
  v8f zx1r[4];
#pragma unroll
  for (int g = 0; g < 4; ++g)
#pragma unroll
    for (int r = 0; r < 8; ++r)
      zx1r[g][r] = zx1[(size_t)(b0 + rowb + r) * G4_ + gm[g] * 16 + nn];

  v8f b2acc[4];
#pragma unroll
  for (int g = 0; g < 4; ++g) {
    float v = b2[gm[g] * 16 + nn];
#pragma unroll
    for (int r = 0; r < 8; ++r) b2acc[g][r] = v;
  }
  v8f boacc;
  {
    float v = (w < 4) ? bout[w * 16 + nn] : 0.f;
#pragma unroll
    for (int r = 0; r < 8; ++r) boacc[r] = v;
  }

  v8f c1 = {}; v8f c2 = {};

  const bf16* Wh1p  = Wh1p0;
  const bf16* Wx2p  = Wx2p0;
  const bf16* Wh2p  = Wh2p0;
  const bf16* Woutp = Woutp0;

  __syncthreads();

#pragma clang loop unroll(disable)
  for (int t = 0; t < T_; ++t) {
    // ---------- layer 1: z1 = zx1 + h1_prev @ Wh1 ----------
    v8f acc[4];
    {
      v16bf a0 = load_a_frag(h1s, lane, 0);
#pragma unroll
      for (int g = 0; g < 4; ++g)          // k = 0 : C = zx1r (no copies)
        acc[g] = wmma_bf16(a0, load_b_frag(Wh1p, lane, 0, gm[g], 64), zx1r[g]);
    }
#pragma unroll
    for (int k = 1; k < 8; ++k) {
      v16bf a = load_a_frag(h1s, lane, k * 32);
#pragma unroll
      for (int g = 0; g < 4; ++g)
        acc[g] = wmma_bf16(a, load_b_frag(Wh1p, lane, k, gm[g], 64), acc[g]);
    }
    v8f h1n;
#pragma unroll
    for (int r = 0; r < 8; ++r) {
      float iv = sigf(acc[0][r]);
      float fv = sigf(acc[1][r]);
      float gv = tanh_(acc[2][r]);
      float ov = sigf(acc[3][r]);
      c1[r]  = fv * c1[r] + iv * gv;
      h1n[r] = ov * tanh_(c1[r]);
    }
    __syncthreads();                               // all done reading h1_prev
#pragma unroll
    for (int r = 0; r < 8; ++r)
      h1s[(rowb + r) * H_ + w * 16 + nn] = (bf16)h1n[r];
    __syncthreads();                               // h1_t visible

    // ---------- layer 2: z2 = b2 + h1_t @ Wx2 + h2_prev @ Wh2 ----------
    {
      v16bf a0 = load_a_frag(h1s, lane, 0);
#pragma unroll
      for (int g = 0; g < 4; ++g)          // k = 0 : C = b2 broadcast
        acc[g] = wmma_bf16(a0, load_b_frag(Wx2p, lane, 0, gm[g], 64), b2acc[g]);
    }
#pragma unroll
    for (int k = 1; k < 8; ++k) {
      v16bf a = load_a_frag(h1s, lane, k * 32);
#pragma unroll
      for (int g = 0; g < 4; ++g)
        acc[g] = wmma_bf16(a, load_b_frag(Wx2p, lane, k, gm[g], 64), acc[g]);
    }
#pragma unroll
    for (int k = 0; k < 8; ++k) {
      v16bf a = load_a_frag(h2s, lane, k * 32);
#pragma unroll
      for (int g = 0; g < 4; ++g)
        acc[g] = wmma_bf16(a, load_b_frag(Wh2p, lane, k, gm[g], 64), acc[g]);
    }
    v8f h2n;
#pragma unroll
    for (int r = 0; r < 8; ++r) {
      float iv = sigf(acc[0][r]);
      float fv = sigf(acc[1][r]);
      float gv = tanh_(acc[2][r]);
      float ov = sigf(acc[3][r]);
      c2[r]  = fv * c2[r] + iv * gv;
      h2n[r] = ov * tanh_(c2[r]);
    }
    __syncthreads();                               // all done reading h2_prev
#pragma unroll
    for (int r = 0; r < 8; ++r)
      h2s[(rowb + r) * H_ + w * 16 + nn] = (bf16)h2n[r];
    __syncthreads();                               // h2_t visible

    // ---------- out_t = h2_t @ W_out + b_out (waves 0..3) ----------
    if (w < 4) {
      v8f oa = wmma_bf16(load_a_frag(h2s, lane, 0),
                         load_b_frag(Woutp, lane, 0, w, 4), boacc);
#pragma unroll
      for (int k = 1; k < 8; ++k)
        oa = wmma_bf16(load_a_frag(h2s, lane, k * 32),
                       load_b_frag(Woutp, lane, k, w, 4), oa);
#pragma unroll
      for (int r = 0; r < 8; ++r)
        out[(size_t)(b0 + rowb + r) * (T_ * F_) + t * F_ + w * 16 + nn] = oa[r];
    }

    // opaque (runtime-zero) stride: defeats LICM of the weight streams
    Wh1p  += wstep;
    Wx2p  += wstep;
    Wh2p  += wstep;
    Woutp += wstep;
  }
}

extern "C" void kernel_launch(void* const* d_in, const int* in_sizes, int n_in,
                              void* d_out, int out_size, void* d_ws, size_t ws_size,
                              hipStream_t stream) {
  const float* inputs = (const float*)d_in[0];
  const float* W_in   = (const float*)d_in[1];
  const float* b_in   = (const float*)d_in[2];
  const float* Wx1    = (const float*)d_in[3];
  const float* Wh1    = (const float*)d_in[4];
  const float* b1     = (const float*)d_in[5];
  const float* Wx2    = (const float*)d_in[6];
  const float* Wh2    = (const float*)d_in[7];
  const float* b2     = (const float*)d_in[8];
  const float* W_out  = (const float*)d_in[9];
  const float* b_out  = (const float*)d_in[10];
  float* out = (float*)d_out;
  (void)in_sizes; (void)n_in; (void)out_size; (void)ws_size;

  // workspace layout (bytes)
  char* ws = (char*)d_ws;
  bf16*  Wh1p  = (bf16*)(ws);                       // 8*64*32*16*2  = 524288
  bf16*  Wx2p  = (bf16*)(ws + 524288);              // 524288
  bf16*  Wh2p  = (bf16*)(ws + 1048576);             // 524288
  bf16*  Woutp = (bf16*)(ws + 1572864);             // 8*4*32*16*2   = 32768
  float* x0    = (float*)(ws + 1605632);            // 256*256*4     = 262144
  float* zx1   = (float*)(ws + 1867776);            // 256*1024*4    = 1048576

  k_x0 <<<(B_ * H_)  / 256, 256, 0, stream>>>(inputs, W_in, b_in, x0);
  k_zx1<<<(B_ * G4_) / 256, 256, 0, stream>>>(x0, Wx1, b1, zx1);
  k_pack<<<(8 * 64 * 32) / 256, 256, 0, stream>>>(Wh1, Wh1p, 64);
  k_pack<<<(8 * 64 * 32) / 256, 256, 0, stream>>>(Wx2, Wx2p, 64);
  k_pack<<<(8 * 64 * 32) / 256, 256, 0, stream>>>(Wh2, Wh2p, 64);
  k_pack<<<(8 * 4 * 32)  / 256, 256, 0, stream>>>(W_out, Woutp, 4);
  lstm_fused<<<16, 512, 0, stream>>>(zx1, b2, Wh1p, Wx2p, Wh2p, Woutp, b_out, out,
                                     (long)0);
}
